// RNNModel_59012850647150
// MI455X (gfx1250) — compile-verified, matrix operand order checked by reading, not run on hardware
//
#include <hip/hip_runtime.h>
#include <hip/hip_bf16.h>
#include <math.h>

// Model dims (match reference)
#define VOCAB  50257
#define EMBED  256
#define HIDDEN 512
#define BATCH  64
#define SEQ    256
#define NWG    16     // workgroups cooperating on the recurrence

typedef __attribute__((ext_vector_type(16))) __bf16 v16bf;
typedef __attribute__((ext_vector_type(8)))  float  v8f;

// One WMMA operand fragment: 16 bf16 per lane (8 VGPRs)
union Frag {
  v16bf          v;
  __bf16         h[16];
  unsigned short u[16];
  uint4          q[2];
};

// Scalar f32 -> bf16 through the compiler's native lowering (v_cvt_* on gfx1250)
__device__ __forceinline__ __bf16 f2bf(float f) { return (__bf16)f; }

// Convert a packed pair of floats to 2 bf16 (4 bytes) using the hardware
// packed converter when the builtin is declared; memcpy makes us agnostic to
// the builtin's exact return type.
__device__ __forceinline__ void cvt_pair(float x, float y, __bf16* dst) {
#if __has_builtin(__builtin_amdgcn_cvt_pk_bf16_f32)
  auto t = __builtin_amdgcn_cvt_pk_bf16_f32(x, y);
  __builtin_memcpy(dst, &t, 4);
#else
  dst[0] = (__bf16)x;
  dst[1] = (__bf16)y;
#endif
}

// Load 8 contiguous f32 (16B-aligned) and convert to 8 bf16
__device__ __forceinline__ void cvt8v(const float* __restrict__ p, __bf16* dst) {
  float4 a = *reinterpret_cast<const float4*>(p);
  float4 b = *reinterpret_cast<const float4*>(p + 4);
  cvt_pair(a.x, a.y, dst + 0);
  cvt_pair(a.z, a.w, dst + 2);
  cvt_pair(b.x, b.y, dst + 4);
  cvt_pair(b.z, b.w, dst + 6);
}

// ---------------------------------------------------------------------------
// Kernel 0: per-launch prep — convert W_hh to bf16, zero h0, reset barrier.
// Must run every launch (harness does not re-poison ws between replays).
// ---------------------------------------------------------------------------
__global__ __launch_bounds__(256) void prep_kernel(const float* __restrict__ W_hh,
                                                   __bf16* __restrict__ whh_bf,
                                                   __bf16* __restrict__ h0,
                                                   int* __restrict__ bar) {
  int idx = blockIdx.x * blockDim.x + threadIdx.x;
  int stride = gridDim.x * blockDim.x;
  // pairwise so the packed converter can be used
  for (int i = idx; i < HIDDEN * HIDDEN / 2; i += stride)
    cvt_pair(W_hh[2 * i], W_hh[2 * i + 1], whh_bf + 2 * i);
  for (int i = idx; i < BATCH * HIDDEN / 2; i += stride)
    reinterpret_cast<unsigned int*>(h0)[i] = 0u;               // h_{-1} = 0
  if (idx == 0) *bar = 0;
}

// ---------------------------------------------------------------------------
// Kernel 1: xproj[s,b,h] = emb[tok[b,s]] @ W_ih^T + b_ih
// GEMM: M = S*B = 16384 (gathered rows), K = E = 256, N = H = 512.
// Block = 8 waves sharing one M-tile, covering 128 columns of H.
// ---------------------------------------------------------------------------
__global__ __launch_bounds__(256) void xproj_kernel(const int*   __restrict__ tokens, // (B,S)
                                                    const float* __restrict__ emb,
                                                    const float* __restrict__ W_ih,   // (H,E)
                                                    const float* __restrict__ b_ih,
                                                    float*       __restrict__ xproj) { // (S*B, H)
  const int lane = threadIdx.x & 31;
  const int wid  = threadIdx.x >> 5;   // 0..7
  const int half = lane >> 4;          // K-half selector per documented A layout
  const int lr   = lane & 15;          // M (for A) / N (for B,C)

  const int m0 = blockIdx.x * 16;                 // tile origin in M = s*BATCH + b
  const int n0 = blockIdx.y * 128 + wid * 16;     // tile origin in H

  // Gather: each lane owns one A row (tiles of 16 never straddle a batch-64 block)
  const int m   = m0 + lr;
  const int s   = m / BATCH;
  const int b   = m % BATCH;
  const int tok = tokens[b * SEQ + s];
  const float* __restrict__ arow = emb  + (long)tok       * EMBED;
  const float* __restrict__ brow = W_ih + (long)(n0 + lr) * EMBED;  // B[k,n] = W_ih[n,k]

  v8f acc = {};
#pragma unroll
  for (int kb = 0; kb < EMBED; kb += 32) {
    Frag A, B;
    cvt8v(arow + kb +      half * 8, A.h);
    cvt8v(arow + kb + 16 + half * 8, A.h + 8);
    cvt8v(brow + kb +      half * 8, B.h);
    cvt8v(brow + kb + 16 + half * 8, B.h + 8);
    acc = __builtin_amdgcn_wmma_f32_16x16x32_bf16(false, A.v, false, B.v,
                                                  (short)0, acc, false, false);
  }

  const int   n    = n0 + lr;
  const float bias = b_ih[n];
#pragma unroll
  for (int v = 0; v < 8; ++v) {
    const int mr = m0 + v + 8 * half;             // C layout: M = v + 8*(lane>=16)
    xproj[(long)mr * HIDDEN + n] = acc[v] + bias;
  }
}

// ---------------------------------------------------------------------------
// Kernel 2: sequential recurrence h_t = tanh(xproj_t + h_{t-1} @ W_hh^T + b_hh)
// Persistent kernel, NWG=16 workgroups; WG g owns columns [32g, 32g+32) with
// its W_hh slice resident in LDS (32 KB) for all 256 steps.  h ping-pongs in a
// bf16 global buffer; steps are separated by a device-scope monotonic barrier.
// ---------------------------------------------------------------------------
__global__ __launch_bounds__(256) void rnn_kernel(const float*   __restrict__ xproj,
                                                  const __bf16*  __restrict__ whh_bf, // (H,H) bf16
                                                  const float*   __restrict__ b_hh,
                                                  __bf16*        __restrict__ hbuf,   // 2 x (B,H) bf16
                                                  int*           __restrict__ bar) {
  __shared__ __align__(16) __bf16 wsl[32 * HIDDEN];   // 32 KB W_hh slice

  const int tid  = threadIdx.x;
  const int lane = tid & 31;
  const int wid  = tid >> 5;
  const int half = lane >> 4;
  const int lr   = lane & 15;
  const int n0   = blockIdx.x * 32;

  // Stage this WG's 32 rows of W_hh (bf16) into LDS, 16B at a time.
  {
    const uint4* src = reinterpret_cast<const uint4*>(whh_bf + (long)n0 * HIDDEN);
    uint4*       dst = reinterpret_cast<uint4*>(wsl);
    for (int i = tid; i < 32 * HIDDEN / 8; i += 256) dst[i] = src[i];
  }
  __syncthreads();

  const int mt   = wid >> 1;            // 4 M-tiles of 16 rows (BATCH = 64)
  const int nt   = wid & 1;             // 2 N-tiles of 16 cols (slice = 32)
  const int m0   = mt * 16;
  const int nloc = nt * 16 + lr;        // local column 0..31
  const int n    = n0 + nloc;
  const float bias = b_hh[n];

  for (int st = 0; st < SEQ; ++st) {
    const __bf16* __restrict__ hprev = hbuf + (st & 1)       * (BATCH * HIDDEN);
    __bf16*       __restrict__ hnext = hbuf + ((st + 1) & 1) * (BATCH * HIDDEN);

    const __bf16* __restrict__ arow = hprev + (long)(m0 + lr) * HIDDEN;
    const __bf16* __restrict__ brow = wsl + nloc * HIDDEN;    // LDS, ds_load_b128

    v8f acc = {};
#pragma unroll 4
    for (int kb = 0; kb < HIDDEN; kb += 32) {
      Frag A, B;
      A.q[0] = *reinterpret_cast<const uint4*>(arow + kb +      half * 8);
      A.q[1] = *reinterpret_cast<const uint4*>(arow + kb + 16 + half * 8);
      B.q[0] = *reinterpret_cast<const uint4*>(brow + kb +      half * 8);
      B.q[1] = *reinterpret_cast<const uint4*>(brow + kb + 16 + half * 8);
      acc = __builtin_amdgcn_wmma_f32_16x16x32_bf16(false, A.v, false, B.v,
                                                    (short)0, acc, false, false);
    }

    const float* __restrict__ xp = xproj + (long)st * (BATCH * HIDDEN);
#pragma unroll
    for (int v = 0; v < 8; ++v) {
      const int mr = m0 + v + 8 * half;
      const float hv = tanhf(acc[v] + xp[(long)mr * HIDDEN + n] + bias);
      hnext[(long)mr * HIDDEN + n] = f2bf(hv);
    }

    // Device-scope barrier across the 16 WGs (monotonic counter, reset by prep)
    __syncthreads();
    if (tid == 0) {
      __threadfence();
      __hip_atomic_fetch_add(bar, 1, __ATOMIC_RELEASE, __HIP_MEMORY_SCOPE_AGENT);
      const int target = NWG * (st + 1);
      while (__hip_atomic_load(bar, __ATOMIC_ACQUIRE, __HIP_MEMORY_SCOPE_AGENT) < target)
        __builtin_amdgcn_s_sleep(1);
    }
    __syncthreads();
  }
}

// ---------------------------------------------------------------------------
// Kernel 3: out = h_last @ W_out^T + b_out   (memory bound on 103 MB W_out)
// M = 64 (4 tiles), K = 512, N = VOCAB; one block covers 32 columns.
// W_out streamed as f32 and converted to bf16 inline (single HBM pass).
// ---------------------------------------------------------------------------
__global__ __launch_bounds__(256) void head_kernel(const __bf16* __restrict__ hlast,  // (B,H) bf16
                                                   const float* __restrict__ W_out,   // (V,H)
                                                   const float* __restrict__ b_out,
                                                   float*       __restrict__ out) {   // (B,V)
  const int lane = threadIdx.x & 31;
  const int wid  = threadIdx.x >> 5;
  const int half = lane >> 4;
  const int lr   = lane & 15;
  const int mt   = wid >> 1;
  const int nt   = wid & 1;
  const int m0   = mt * 16;
  const int n    = blockIdx.x * 32 + nt * 16 + lr;
  const int nr   = (n < VOCAB) ? n : (VOCAB - 1);  // clamp loads; stores guarded

  const __bf16* __restrict__ arow = hlast + (long)(m0 + lr) * HIDDEN;
  const float*  __restrict__ brow = W_out + (long)nr * HIDDEN;

  v8f acc = {};
#pragma unroll 4
  for (int kb = 0; kb < HIDDEN; kb += 32) {
    Frag A, B;
    A.q[0] = *reinterpret_cast<const uint4*>(arow + kb +      half * 8);
    A.q[1] = *reinterpret_cast<const uint4*>(arow + kb + 16 + half * 8);
    cvt8v(brow + kb +      half * 8, B.h);
    cvt8v(brow + kb + 16 + half * 8, B.h + 8);
    __builtin_prefetch(brow + kb + 128, 0, 1);     // global_prefetch_b8 ahead in K
    acc = __builtin_amdgcn_wmma_f32_16x16x32_bf16(false, A.v, false, B.v,
                                                  (short)0, acc, false, false);
  }

  if (n < VOCAB) {
    const float bias = b_out[n];
#pragma unroll
    for (int v = 0; v < 8; ++v) {
      const int mr = m0 + v + 8 * half;
      out[(long)mr * VOCAB + n] = acc[v] + bias;
    }
  }
}

// ---------------------------------------------------------------------------
// Workspace layout (bytes):
//   [0, 32 MB)          xproj f32  (S*B, H)       = 33,554,432
//   [+0.5 MB)           W_hh bf16  (H, H)         =    524,288
//   [+128 KB)           h ping/pong bf16 2x(B,H)  =    131,072
//   [+256 B)            barrier counter
// ---------------------------------------------------------------------------
extern "C" void kernel_launch(void* const* d_in, const int* in_sizes, int n_in,
                              void* d_out, int out_size, void* d_ws, size_t ws_size,
                              hipStream_t stream) {
  (void)in_sizes; (void)n_in; (void)out_size; (void)ws_size;
  const int*   tokens = (const int*)  d_in[0];
  const float* emb    = (const float*)d_in[1];
  const float* W_ih   = (const float*)d_in[2];
  const float* b_ih   = (const float*)d_in[3];
  const float* W_hh   = (const float*)d_in[4];
  const float* b_hh   = (const float*)d_in[5];
  const float* W_out  = (const float*)d_in[6];
  const float* b_out  = (const float*)d_in[7];
  float* out = (float*)d_out;

  char* ws = (char*)d_ws;
  float*   xproj  = (float*) (ws);
  __bf16*  whh_bf = (__bf16*)(ws + 33554432);
  __bf16*  hbuf   = (__bf16*)(ws + 34078720);
  int*     bar    = (int*)   (ws + 34209792);

  prep_kernel<<<512, 256, 0, stream>>>(W_hh, whh_bf, hbuf, bar);

  dim3 g1(SEQ * BATCH / 16, HIDDEN / 128);   // 1024 x 4 blocks
  xproj_kernel<<<g1, 256, 0, stream>>>(tokens, emb, W_ih, b_ih, xproj);

  rnn_kernel<<<NWG, 256, 0, stream>>>(xproj, whh_bf, b_hh, hbuf, bar);

  // SEQ is even -> final hidden state lands back in buffer 0
  head_kernel<<<(VOCAB + 31) / 32, 256, 0, stream>>>(hbuf, W_out, b_out, out);
}
